// _NonLocalNdGc_36240934043974
// MI455X (gfx1250) — matrix operand order repaired and checked by reading, not compiled, not run on hardware
//
#include <hip/hip_runtime.h>
#include <hip/hip_bf16.h>

typedef __bf16 bf16;
typedef __attribute__((ext_vector_type(16))) __bf16 bf16x16;
typedef __attribute__((ext_vector_type(8)))  __bf16 bf16x8;
typedef __attribute__((ext_vector_type(8)))  float  v8f;

__device__ inline bf16x16 cat8(bf16x8 lo, bf16x8 hi) {
  return __builtin_shufflevector(lo, hi, 0,1,2,3,4,5,6,7,8,9,10,11,12,13,14,15);
}

__device__ inline v8f wmma_bf16(bf16x16 a, bf16x16 b, v8f c) {
  // D = A(16x32) * B(32x16) + C, f32 accumulate
  return __builtin_amdgcn_wmma_f32_16x16x32_bf16(false, a, false, b, (short)0, c, false, false);
}

// ---------------------------------------------------------------------------
// f32 -> bf16 convert (weights)
// ---------------------------------------------------------------------------
__global__ void f32_to_bf16(const float* __restrict__ src, bf16* __restrict__ dst, int n) {
  int i = blockIdx.x * blockDim.x + threadIdx.x;
  if (i < n) dst[i] = (bf16)src[i];
}

// ---------------------------------------------------------------------------
// x [B,512,4096] f32  ->  xT [B,4096,512] bf16   (transpose + convert)
// ---------------------------------------------------------------------------
__global__ __launch_bounds__(256)
void transpose_convert(const float* __restrict__ x, bf16* __restrict__ xT) {
  __shared__ float tile[32][33];
  const int b  = blockIdx.z;
  const int c0 = blockIdx.y * 32;
  const int n0 = blockIdx.x * 32;
  const int tx = threadIdx.x & 31;
  const int ty = threadIdx.x >> 5;   // 0..7
  const long xb = (long)b * 512 * 4096;
#pragma unroll
  for (int i = 0; i < 4; ++i) {
    int c = ty + i * 8;
    tile[c][tx] = x[xb + (long)(c0 + c) * 4096 + n0 + tx];
  }
  __syncthreads();
#pragma unroll
  for (int i = 0; i < 4; ++i) {
    int n = ty + i * 8;
    xT[(long)b * 4096 * 512 + (long)(n0 + n) * 512 + c0 + tx] = (bf16)tile[tx][n];
  }
}

// ---------------------------------------------------------------------------
// maxpool2x2 of x then transpose: xdT [B,1024,512] bf16
// ---------------------------------------------------------------------------
__global__ __launch_bounds__(256)
void pool_transpose(const float* __restrict__ x, bf16* __restrict__ xdT) {
  __shared__ float tile[32][33];
  const int b  = blockIdx.z;
  const int c0 = blockIdx.y * 32;
  const int m0 = blockIdx.x * 32;
  const int tx = threadIdx.x & 31;
  const int ty = threadIdx.x >> 5;
#pragma unroll
  for (int i = 0; i < 4; ++i) {
    int cc = ty + i * 8;
    int c  = c0 + cc;
    int m  = m0 + tx;
    int mh = m >> 5, mw = m & 31;            // pooled grid is 32x32
    const float* p = x + (long)(b * 512 + c) * 4096 + (mh * 2) * 64 + mw * 2;
    float v0 = p[0], v1 = p[1], v2 = p[64], v3 = p[65];
    tile[cc][tx] = fmaxf(fmaxf(v0, v1), fmaxf(v2, v3));
  }
  __syncthreads();
#pragma unroll
  for (int i = 0; i < 4; ++i) {
    int mm = ty + i * 8;
    xdT[(long)b * 1024 * 512 + (long)(m0 + mm) * 512 + c0 + tx] = (bf16)tile[tx][mm];
  }
}

// ---------------------------------------------------------------------------
// Generic Out[M,N] = A[M,512] * Bsrc[N,512]^T  (+bias[N]) * outScale, bf16 out.
// Wave tile 32x64; block = 8 waves = 64 rows x 256 cols. grid.z = batch.
// ---------------------------------------------------------------------------
__global__ __launch_bounds__(256)
void gemm_bf16_nt(const bf16* __restrict__ A, const bf16* __restrict__ Bsrc,
                  const float* __restrict__ bias, bf16* __restrict__ Out,
                  int ldo, float outScale, long aStrideZ, long bStrideZ, long oStrideZ) {
  const int lane = threadIdx.x & 31;
  const int wave = threadIdx.x >> 5;
  const int r = lane & 15, h = lane >> 4;
  const int waveRow = wave >> 2, waveCol = wave & 3;
  const long zb = blockIdx.z;
  const bf16* Ab = A + zb * aStrideZ;
  const bf16* Bb = Bsrc + zb * bStrideZ;
  bf16* Ob = Out + zb * oStrideZ;
  const int row0 = blockIdx.y * 64 + waveRow * 32;
  const int col0 = blockIdx.x * 256 + waveCol * 64;

  v8f acc[2][4];
  const v8f vz = {};
#pragma unroll
  for (int i = 0; i < 2; ++i)
#pragma unroll
    for (int j = 0; j < 4; ++j) acc[i][j] = vz;

  for (int kc = 0; kc < 16; ++kc) {          // K = 512 in chunks of 32
    const int klo = kc * 32 + h * 16;
    bf16x16 a[2];
#pragma unroll
    for (int rt = 0; rt < 2; ++rt) {
      const bf16* p = Ab + (long)(row0 + rt * 16 + r) * 512;
      a[rt] = cat8(*(const bf16x8*)(p + klo), *(const bf16x8*)(p + klo + 8));
    }
#pragma unroll
    for (int ct = 0; ct < 4; ++ct) {
      const bf16* p = Bb + (long)(col0 + ct * 16 + r) * 512 + kc * 32 + h * 16;
      bf16x16 bfrag = cat8(*(const bf16x8*)p, *(const bf16x8*)(p + 8));
#pragma unroll
      for (int rt = 0; rt < 2; ++rt)
        acc[rt][ct] = wmma_bf16(a[rt], bfrag, acc[rt][ct]);
    }
  }
#pragma unroll
  for (int ct = 0; ct < 4; ++ct) {
    const int col = col0 + ct * 16 + r;
    const float bv = bias ? bias[col] : 0.0f;
#pragma unroll
    for (int rt = 0; rt < 2; ++rt)
#pragma unroll
      for (int j = 0; j < 8; ++j) {
        const int row = row0 + rt * 16 + j + 8 * h;
        Ob[(long)row * ldo + col] = (bf16)((acc[rt][ct][j] + bv) * outScale);
      }
  }
}

// ---------------------------------------------------------------------------
// mask[b,1024] = softmax_j( Wm . xdT[b,j,:] + bm )
// ---------------------------------------------------------------------------
__global__ __launch_bounds__(256)
void mask_kernel(const bf16* __restrict__ xdT, const float* __restrict__ Wm,
                 const float* __restrict__ bm, float* __restrict__ mask) {
  __shared__ float w[512];
  __shared__ float vals[1024];
  __shared__ float red[256];
  const int b = blockIdx.x;
  const int t = threadIdx.x;
  for (int c = t; c < 512; c += 256) w[c] = Wm[c];
  __syncthreads();
  const float bias = bm[0];
  for (int jj = 0; jj < 4; ++jj) {
    int j = t + jj * 256;
    const bf16* row = xdT + (long)b * 1024 * 512 + (long)j * 512;
    float acc = bias;
    for (int c = 0; c < 512; ++c) acc += (float)row[c] * w[c];
    vals[j] = acc;
  }
  __syncthreads();
  float mx = -1e30f;
  for (int jj = 0; jj < 4; ++jj) mx = fmaxf(mx, vals[t + jj * 256]);
  red[t] = mx; __syncthreads();
  for (int s = 128; s > 0; s >>= 1) { if (t < s) red[t] = fmaxf(red[t], red[t + s]); __syncthreads(); }
  mx = red[0]; __syncthreads();
  float sm = 0.f;
  for (int jj = 0; jj < 4; ++jj) { float e = __expf(vals[t + jj * 256] - mx); vals[t + jj * 256] = e; sm += e; }
  __syncthreads();
  red[t] = sm; __syncthreads();
  for (int s = 128; s > 0; s >>= 1) { if (t < s) red[t] += red[t + s]; __syncthreads(); }
  const float inv = 1.0f / red[0];
  for (int jj = 0; jj < 4; ++jj) mask[(long)b * 1024 + t + jj * 256] = vals[t + jj * 256] * inv;
}

// ---------------------------------------------------------------------------
// gc[b,c] = sum_m v[b,c,m] * mask[b,m]
// ---------------------------------------------------------------------------
__global__ __launch_bounds__(256)
void gc_kernel(const bf16* __restrict__ v, const float* __restrict__ mask,
               float* __restrict__ gc) {
  __shared__ float mk[1024];
  const int b = blockIdx.x;
  const int t = threadIdx.x;
  for (int m = t; m < 1024; m += 256) mk[m] = mask[(long)b * 1024 + m];
  __syncthreads();
  for (int ci = 0; ci < 2; ++ci) {
    int c = t + ci * 256;
    const bf16* row = v + (long)(b * 512 + c) * 1024;
    float acc = 0.f;
    for (int m = 0; m < 1024; ++m) acc += (float)row[m] * mk[m];
    gc[b * 512 + c] = acc;
  }
}

// ---------------------------------------------------------------------------
// Flash-style attention with online softmax.
// One wave owns (batch b, 16 queries n0.., 128 channels c0..).
// qT already scaled by 1/16 (SCALE). Epilogue fuses x + gamma*out_sim + gc.
// ---------------------------------------------------------------------------
__global__ __launch_bounds__(256)
void attn_kernel(const float* __restrict__ x, const bf16* __restrict__ qT,
                 const bf16* __restrict__ kT, const bf16* __restrict__ v,
                 const float* __restrict__ gc, const float* __restrict__ gamma,
                 float* __restrict__ out) {
  __shared__ __align__(16) bf16 pShm[8][16 * 32];   // per-wave P staging (1KB each)
  __shared__ float statShm[8][16];                  // per-wave row stats
  const int lane = threadIdx.x & 31;
  const int wave = threadIdx.x >> 5;
  const int gw = blockIdx.x * 8 + wave;             // 0..8191
  const int b = gw >> 10;
  const int rem = gw & 1023;
  const int n0 = (rem >> 2) * 16;                   // query tile base
  const int c0 = (rem & 3) * 128;                   // channel chunk base
  const int r = lane & 15, h = lane >> 4;

  // resident q A-fragments: 16 queries x 256 P, 8 chunks of K=32
  bf16x16 qa[8];
  const bf16* qrow = qT + ((long)b * 4096 + n0 + r) * 256;
#pragma unroll
  for (int kc = 0; kc < 8; ++kc) {
    int klo = kc * 32 + h * 16;
    qa[kc] = cat8(*(const bf16x8*)(qrow + klo), *(const bf16x8*)(qrow + klo + 8));
  }

  const v8f vz = {};
  v8f oacc[8];
#pragma unroll
  for (int i = 0; i < 8; ++i) oacc[i] = vz;
  float mrun[8], lrun[8];
#pragma unroll
  for (int j = 0; j < 8; ++j) { mrun[j] = -1e30f; lrun[j] = 0.0f; }

  bf16* pw = &pShm[wave][0];
  float* st = &statShm[wave][0];
  const bf16* kTb = kT + (long)b * 1024 * 256;
  const bf16* vb  = v  + (long)b * 512 * 1024;

  for (int kb = 0; kb < 32; ++kb) {
    const int m0 = kb * 32;
    if (kb + 1 < 32) __builtin_prefetch(kTb + (long)(m0 + 32 + r) * 256, 0, 0);

    // S tiles: queries x 32 keys (two 16-key tiles), K = 256 over P
    v8f s0 = vz, s1 = vz;
    {
      const bf16* k0 = kTb + (long)(m0 + r) * 256 + h * 16;
      const bf16* k1 = kTb + (long)(m0 + 16 + r) * 256 + h * 16;
#pragma unroll
      for (int kc = 0; kc < 8; ++kc) {
        const bf16* p0 = k0 + kc * 32;
        const bf16* p1 = k1 + kc * 32;
        bf16x16 bk0 = cat8(*(const bf16x8*)p0, *(const bf16x8*)(p0 + 8));
        bf16x16 bk1 = cat8(*(const bf16x8*)p1, *(const bf16x8*)(p1 + 8));
        s0 = wmma_bf16(qa[kc], bk0, s0);
        s1 = wmma_bf16(qa[kc], bk1, s1);
      }
    }

    // online softmax, per query row j (+8h)
    float scale[8];
#pragma unroll
    for (int j = 0; j < 8; ++j) {
      float bmax = fmaxf(s0[j], s1[j]);
#pragma unroll
      for (int d = 1; d < 16; d <<= 1) bmax = fmaxf(bmax, __shfl_xor(bmax, d, 32));
      float mn = fmaxf(mrun[j], bmax);
      float sc = __expf(mrun[j] - mn);
      float p0 = __expf(s0[j] - mn);
      float p1 = __expf(s1[j] - mn);
      float rs = p0 + p1;
#pragma unroll
      for (int d = 1; d < 16; d <<= 1) rs += __shfl_xor(rs, d, 32);
      lrun[j] = lrun[j] * sc + rs;
      mrun[j] = mn;
      scale[j] = sc;
      // P row-major [query 16][key 32] in LDS
      pw[(j + 8 * h) * 32 + r]      = (bf16)p0;
      pw[(j + 8 * h) * 32 + 16 + r] = (bf16)p1;
    }
    if (r == 0) {
#pragma unroll
      for (int j = 0; j < 8; ++j) st[8 * h + j] = scale[j];
    }
    asm volatile("s_wait_dscnt 0" ::: "memory");

    // rescale O by per-query factor (query = column of O tiles)
    const float colScale = st[r];
#pragma unroll
    for (int i = 0; i < 8; ++i)
#pragma unroll
      for (int j = 0; j < 8; ++j) oacc[i][j] *= colScale;

    // B fragment of P^T: B[kk, q] = P[q, kk]
    const bf16* pl = pw + r * 32 + h * 16;
    bf16x16 bp = cat8(*(const bf16x8*)pl, *(const bf16x8*)(pl + 8));

    // O[c,q] += sum_kk v[c, m0+kk] * P[q, kk]  -- 8 channel row-tiles
#pragma unroll
    for (int rt = 0; rt < 8; ++rt) {
      const bf16* vrow = vb + (long)(c0 + rt * 16 + r) * 1024 + m0;
      bf16x16 av = cat8(*(const bf16x8*)(vrow + h * 16), *(const bf16x8*)(vrow + h * 16 + 8));
      oacc[rt] = wmma_bf16(av, bp, oacc[rt]);
    }
  }

  // broadcast 1/l per query via stats
  if (r == 0) {
#pragma unroll
    for (int j = 0; j < 8; ++j) st[8 * h + j] = lrun[j];
  }
  asm volatile("s_wait_dscnt 0" ::: "memory");
  const float linv = 1.0f / st[r];
  const float g = gamma[0];

#pragma unroll
  for (int rt = 0; rt < 8; ++rt)
#pragma unroll
    for (int j = 0; j < 8; ++j) {
      const int c = c0 + rt * 16 + j + 8 * h;
      const long idx = (long)(b * 512 + c) * 4096 + n0 + r;
      out[idx] = x[idx] + g * (oacc[rt][j] * linv) + gc[b * 512 + c];
    }
}

// ---------------------------------------------------------------------------
extern "C" void kernel_launch(void* const* d_in, const int* in_sizes, int n_in,
                              void* d_out, int out_size, void* d_ws, size_t ws_size,
                              hipStream_t stream) {
  (void)in_sizes; (void)n_in; (void)out_size; (void)ws_size;
  const float* x     = (const float*)d_in[0];
  const float* Wq    = (const float*)d_in[1];
  const float* bq    = (const float*)d_in[2];
  const float* Wk    = (const float*)d_in[3];
  const float* bk    = (const float*)d_in[4];
  const float* Wm    = (const float*)d_in[5];
  const float* bm    = (const float*)d_in[6];
  const float* Wv    = (const float*)d_in[7];
  const float* gamma = (const float*)d_in[8];
  float* out = (float*)d_out;

  char* ws = (char*)d_ws;
  size_t off = 0;
  auto alloc = [&](size_t bytes) -> void* {
    void* p = ws + off; off += (bytes + 255) & ~(size_t)255; return p;
  };
  bf16* xT   = (bf16*)alloc((size_t)8 * 4096 * 512 * 2);
  bf16* xdT  = (bf16*)alloc((size_t)8 * 1024 * 512 * 2);
  bf16* WqB  = (bf16*)alloc((size_t)256 * 512 * 2);
  bf16* WkB  = (bf16*)alloc((size_t)256 * 512 * 2);
  bf16* WvB  = (bf16*)alloc((size_t)512 * 512 * 2);
  bf16* qT   = (bf16*)alloc((size_t)8 * 4096 * 256 * 2);
  bf16* kT   = (bf16*)alloc((size_t)8 * 1024 * 256 * 2);
  bf16* vB   = (bf16*)alloc((size_t)8 * 512 * 1024 * 2);
  float* mask = (float*)alloc((size_t)8 * 1024 * 4);
  float* gc   = (float*)alloc((size_t)8 * 512 * 4);

  f32_to_bf16<<<(131072 + 255) / 256, 256, 0, stream>>>(Wq, WqB, 131072);
  f32_to_bf16<<<(131072 + 255) / 256, 256, 0, stream>>>(Wk, WkB, 131072);
  f32_to_bf16<<<(262144 + 255) / 256, 256, 0, stream>>>(Wv, WvB, 262144);

  transpose_convert<<<dim3(128, 16, 8), 256, 0, stream>>>(x, xT);
  pool_transpose<<<dim3(32, 16, 8), 256, 0, stream>>>(x, xdT);

  // qT[n,p] = (xT . Wq^T + bq)/16   M=32768 N=256
  gemm_bf16_nt<<<dim3(1, 512, 1), 256, 0, stream>>>(xT, WqB, bq, qT, 256,
                                                    1.0f / 16.0f, 0, 0, 0);
  // kT[m,p] = xdT . Wk^T + bk       M=8192 N=256
  gemm_bf16_nt<<<dim3(1, 128, 1), 256, 0, stream>>>(xdT, WkB, bk, kT, 256,
                                                    1.0f, 0, 0, 0);
  // v[b][c,m] = Wv . xd[b]          M=512 N=1024 per batch
  gemm_bf16_nt<<<dim3(4, 8, 8), 256, 0, stream>>>(WvB, xdT, nullptr, vB, 1024, 1.0f,
                                                  0, (long)1024 * 512, (long)512 * 1024);

  mask_kernel<<<8, 256, 0, stream>>>(xdT, Wm, bm, mask);
  gc_kernel<<<8, 256, 0, stream>>>(vB, mask, gc);

  attn_kernel<<<1024, 256, 0, stream>>>(x, qT, kT, vB, gc, gamma, out);
}
